// TransformerLayer_48069273977135
// MI455X (gfx1250) — matrix operand order repaired and checked by reading, not compiled, not run on hardware
//
#include <hip/hip_runtime.h>

// ---------------------------------------------------------------------------
// Types for CDNA5 WMMA (wave32, 16x16x32 f16 -> f32)
// ---------------------------------------------------------------------------
typedef __attribute__((ext_vector_type(16))) _Float16 v16h;
typedef __attribute__((ext_vector_type(8)))  _Float16 v8h;
typedef __attribute__((ext_vector_type(8)))  float    v8f;

__device__ __forceinline__ v8f wmma16(v16h a, v16h b, v8f c) {
    return __builtin_amdgcn_wmma_f32_16x16x32_f16(false, a, false, b, (short)0, c, false, false);
}

__device__ __forceinline__ v16h mk16(v8h lo, v8h hi) {
    v16h r;
#pragma unroll
    for (int i = 0; i < 8; ++i) { r[i] = lo[i]; r[i + 8] = hi[i]; }
    return r;
}

// A fragment (16x32 f16), row-major source, row stride ld (halves).
// lanes 0-15: row M=lane, K chunks [0..7],[16..23]; lanes 16-31: row M=lane-16, [8..15],[24..31]
__device__ __forceinline__ v16h loadA16(const _Float16* base, int ld, int lane) {
    int m  = lane & 15;
    int ko = (lane < 16) ? 0 : 8;
    const _Float16* p = base + (size_t)m * ld + ko;
    return mk16(*(const v8h*)p, *(const v8h*)(p + 16));
}

// B fragment (32x16 f16) from N-major ("transposed") storage, row stride ld:
// lane<16: col N=lane, K [0..7],[16..23]; lane>=16: col N=lane-16, K [8..15],[24..31]
__device__ __forceinline__ v16h loadBt16(const _Float16* base, int ld, int lane) {
    int n  = lane & 15;
    int ko = (lane < 16) ? 0 : 8;
    const _Float16* p = base + (size_t)n * ld + ko;
    return mk16(*(const v8h*)p, *(const v8h*)(p + 16));
}

// ---------------------------------------------------------------------------
// Model constants
// ---------------------------------------------------------------------------
#define NB   16
#define NH_  12
#define DD   384
#define FF_  1536
#define HW_  56
#define TOK  50176          // 16*56*56
#define SCALE_ 0.17677669529663687f   // 1/sqrt(32)

// ---------------------------------------------------------------------------
// f32 -> f16 convert
// ---------------------------------------------------------------------------
__global__ __launch_bounds__(256) void cvt_kernel(const float* s, _Float16* d, int n) {
    int i = blockIdx.x * 256 + threadIdx.x;
    if (i < n) d[i] = (_Float16)s[i];
}

// ---------------------------------------------------------------------------
// LayerNorm (fp32 in -> f16 out). One wave per token, D=384 = 12 * 32 lanes.
// ---------------------------------------------------------------------------
__global__ __launch_bounds__(256) void ln_kernel(const float* __restrict__ src,
                                                 const float* __restrict__ g,
                                                 const float* __restrict__ be,
                                                 _Float16* __restrict__ dst, int ntok) {
    int wv = threadIdx.x >> 5, lane = threadIdx.x & 31;
    int tok = blockIdx.x * 8 + wv;
    if (tok >= ntok) return;
    const float* p = src + (size_t)tok * DD;
    float v[12];
    float s = 0.f;
#pragma unroll
    for (int i = 0; i < 12; ++i) { v[i] = p[lane + 32 * i]; s += v[i]; }
#pragma unroll
    for (int off = 1; off < 32; off <<= 1) s += __shfl_xor(s, off, 32);
    float mean = s * (1.f / DD);
    float var = 0.f;
#pragma unroll
    for (int i = 0; i < 12; ++i) { float d0 = v[i] - mean; var += d0 * d0; }
#pragma unroll
    for (int off = 1; off < 32; off <<= 1) var += __shfl_xor(var, off, 32);
    float rstd = rsqrtf(var * (1.f / DD) + 1e-6f);
    _Float16* o = dst + (size_t)tok * DD;
#pragma unroll
    for (int i = 0; i < 12; ++i) {
        int c = lane + 32 * i;
        o[c] = (_Float16)((v[i] - mean) * rstd * g[c] + be[c]);
    }
}

// ---------------------------------------------------------------------------
// WMMA GEMM: C[M,N] = act(A[M,K](f16) * B[K,N](f16) + bias) (+ resid)
// 256 threads = 8 waves; block tile 128x128, BK=32, double-buffered LDS.
// Waves 4(m) x 2(n); wave tile 32x64 -> 8 WMMA per K-step.
// M arbitrary (A rows clamped, stores predicated); N%128==0; K%32==0.
// ---------------------------------------------------------------------------
template <bool SILU>
__global__ __launch_bounds__(256) void gemm_kernel(const _Float16* __restrict__ A,
                                                   const _Float16* __restrict__ B,
                                                   const float* __restrict__ bias,
                                                   const float* __restrict__ resid,
                                                   float* __restrict__ outF,
                                                   _Float16* __restrict__ outH,
                                                   int M, int N, int K) {
    __shared__ __align__(16) _Float16 As[2][128 * 32];
    __shared__ __align__(16) _Float16 Bt[2][128 * 32];   // N-major: Bt[n][k]

    int bn = blockIdx.x * 128;
    int bm = blockIdx.y * 128;
    int tid = threadIdx.x, lane = tid & 31, wv = tid >> 5;
    int wm = wv >> 1, wn = wv & 1;

    // thread's fixed staging coordinates
    int ar = tid >> 1, ac = (tid & 1) * 16;        // A: 128 rows x 32 halves
    int agr = bm + ar; if (agr >= M) agr = M - 1;  // clamp (stores predicated)
    const _Float16* aBase = A + (size_t)agr * K + ac;
    int bkr = tid >> 3, bc0 = (tid & 7) * 16;      // B: 32 rows x 128 cols
    const _Float16* bBase = B + (size_t)bkr * N + bn + bc0;

    auto stage = [&](int buf, int kb) {
        const _Float16* ap = aBase + kb;
        *(v8h*)&As[buf][ar * 32 + ac]     = *(const v8h*)ap;
        *(v8h*)&As[buf][ar * 32 + ac + 8] = *(const v8h*)(ap + 8);
        __builtin_prefetch(ap + 64, 0, 1);         // pull K+64 tile toward WGP
        const _Float16* bp = bBase + (size_t)kb * N;
        v8h b0 = *(const v8h*)bp, b1 = *(const v8h*)(bp + 8);
#pragma unroll
        for (int j = 0; j < 8; ++j) {
            Bt[buf][(bc0 + j) * 32 + bkr]     = b0[j];
            Bt[buf][(bc0 + 8 + j) * 32 + bkr] = b1[j];
        }
    };

    v8f acc[8];
#pragma unroll
    for (int t = 0; t < 8; ++t)
#pragma unroll
        for (int i = 0; i < 8; ++i) acc[t][i] = 0.f;

    int nk = K >> 5;
    stage(0, 0);
    for (int i = 0; i < nk; ++i) {
        __syncthreads();
        if (i + 1 < nk) stage((i + 1) & 1, (i + 1) * 32);

        const _Float16* Ab = As[i & 1] + (wm * 32) * 32;
        const _Float16* Bb = Bt[i & 1] + (wn * 64) * 32;
        v16h a0 = loadA16(Ab, 32, lane);
        v16h a1 = loadA16(Ab + 16 * 32, 32, lane);
#pragma unroll
        for (int nj = 0; nj < 4; ++nj) {
            v16h bb = loadBt16(Bb + nj * 16 * 32, 32, lane);
            acc[nj]     = wmma16(a0, bb, acc[nj]);
            acc[4 + nj] = wmma16(a1, bb, acc[4 + nj]);
        }
    }

    // Epilogue: C layout -> lane holds (m = r + (lane>=16?8:0), n = lane&15) per tile
#pragma unroll
    for (int mi = 0; mi < 2; ++mi)
#pragma unroll
        for (int nj = 0; nj < 4; ++nj) {
            v8f c = acc[mi * 4 + nj];
            int gn  = bn + wn * 64 + nj * 16 + (lane & 15);
            int gm0 = bm + wm * 32 + mi * 16 + ((lane < 16) ? 0 : 8);
            float bv = bias ? bias[gn] : 0.f;
#pragma unroll
            for (int r = 0; r < 8; ++r) {
                int gm = gm0 + r;
                if (gm < M) {
                    float v = c[r] + bv;
                    if (SILU) v = v / (1.f + __expf(-v));
                    if (resid) v += resid[(size_t)gm * N + gn];
                    if (outF) outF[(size_t)gm * N + gn] = v;
                    if (outH) outH[(size_t)gm * N + gn] = (_Float16)v;
                }
            }
        }
}

// ---------------------------------------------------------------------------
// Local window attention. One block = (b, window, head); 4 waves, 16 queries
// each. qkv layout [token, 1152] : q | k | v each NH*32.
// ---------------------------------------------------------------------------
__global__ __launch_bounds__(128) void local_attn_kernel(const _Float16* __restrict__ qkv,
                                                         _Float16* __restrict__ out) {
    __shared__ __align__(16) _Float16 Vt[32 * 64];     // [dh][key]
    __shared__ __align__(16) _Float16 Plds[4 * 16 * 64];

    int bid = blockIdx.x;
    int head = bid % NH_;
    int win  = (bid / NH_) % 49;
    int b    = bid / (NH_ * 49);
    int wy = win / 7, wx = win % 7;
    int tid = threadIdx.x, lane = tid & 31, wv = tid >> 5;

    auto tok = [&](int q) { return (b * HW_ + wy * 8 + (q >> 3)) * HW_ + wx * 8 + (q & 7); };

    // stage V transposed: Vt[dh][key]
    {
        int r = tid >> 1;                // key 0..63
        int c0 = (tid & 1) * 16;         // dh chunk
        const _Float16* vp = qkv + (size_t)tok(r) * 1152 + 768 + head * 32 + c0;
        v8h a = *(const v8h*)vp, bb = *(const v8h*)(vp + 8);
#pragma unroll
        for (int j = 0; j < 8; ++j) {
            Vt[(c0 + j) * 64 + r]     = a[j];
            Vt[(c0 + 8 + j) * 64 + r] = bb[j];
        }
    }
    __syncthreads();

    int q0 = wv * 16;
    int m = lane & 15;
    int koA = (lane < 16) ? 0 : 8;

    // Q fragment straight from global
    const _Float16* qp = qkv + (size_t)tok(q0 + m) * 1152 + head * 32 + koA;
    v16h aQ = mk16(*(const v8h*)qp, *(const v8h*)(qp + 16));

    v8f s[4];
#pragma unroll
    for (int t = 0; t < 4; ++t)
#pragma unroll
        for (int i = 0; i < 8; ++i) s[t][i] = 0.f;

    // S = Q K^T : K rows are natural B-fragments (lane = key column)
    int koB = (lane < 16) ? 0 : 8;
#pragma unroll
    for (int t = 0; t < 4; ++t) {
        int key = t * 16 + (lane & 15);
        const _Float16* kp = qkv + (size_t)tok(key) * 1152 + 384 + head * 32 + koB;
        v16h bK = mk16(*(const v8h*)kp, *(const v8h*)(kp + 16));
        s[t] = wmma16(aQ, bK, s[t]);
    }

    // softmax over 64 keys (rows split across half-waves; xor masks <16 stay in half)
#pragma unroll
    for (int r = 0; r < 8; ++r) {
        float mx = -1e30f;
#pragma unroll
        for (int t = 0; t < 4; ++t) { float v = s[t][r] * SCALE_; s[t][r] = v; mx = fmaxf(mx, v); }
#pragma unroll
        for (int off = 1; off < 16; off <<= 1) mx = fmaxf(mx, __shfl_xor(mx, off, 32));
        float sum = 0.f;
#pragma unroll
        for (int t = 0; t < 4; ++t) { float e = __expf(s[t][r] - mx); s[t][r] = e; sum += e; }
#pragma unroll
        for (int off = 1; off < 16; off <<= 1) sum += __shfl_xor(sum, off, 32);
        float inv = 1.f / sum;
        int rowl = r + ((lane < 16) ? 0 : 8);
#pragma unroll
        for (int t = 0; t < 4; ++t)
            Plds[wv * 1024 + rowl * 64 + t * 16 + (lane & 15)] = (_Float16)(s[t][r] * inv);
    }

    // O = P V
    v8f o[2];
#pragma unroll
    for (int j = 0; j < 2; ++j)
#pragma unroll
        for (int i = 0; i < 8; ++i) o[j][i] = 0.f;
#pragma unroll
    for (int kc = 0; kc < 2; ++kc) {
        const _Float16* pp = Plds + wv * 1024 + m * 64 + kc * 32 + koA;
        v16h aP = mk16(*(const v8h*)pp, *(const v8h*)(pp + 16));
#pragma unroll
        for (int j = 0; j < 2; ++j) {
            int n = j * 16 + (lane & 15);
            const _Float16* vp = Vt + n * 64 + kc * 32 + koB;
            v16h bV = mk16(*(const v8h*)vp, *(const v8h*)(vp + 16));
            o[j] = wmma16(aP, bV, o[j]);
        }
    }
#pragma unroll
    for (int j = 0; j < 2; ++j)
#pragma unroll
        for (int r = 0; r < 8; ++r) {
            int q = q0 + r + ((lane < 16) ? 0 : 8);
            out[(size_t)tok(q) * DD + head * 32 + j * 16 + (lane & 15)] = (_Float16)(o[j][r]);
        }
}

// ---------------------------------------------------------------------------
// Global attention: 49 reduced K/V tokens (padded to 64), 64 queries/block.
// q: [50176,384]; kv: [784, 768] (k | v, each NH*32). One block = (b,head,qtile)
// ---------------------------------------------------------------------------
__global__ __launch_bounds__(128) void global_attn_kernel(const _Float16* __restrict__ qbuf,
                                                          const _Float16* __restrict__ kvbuf,
                                                          _Float16* __restrict__ out) {
    __shared__ __align__(16) _Float16 Vt[32 * 64];
    __shared__ __align__(16) _Float16 Plds[4 * 16 * 64];

    int bid = blockIdx.x;
    int qt   = bid % 49;
    int head = (bid / 49) % NH_;
    int b    = bid / (49 * NH_);
    int qbase = b * 3136 + qt * 64;
    int tid = threadIdx.x, lane = tid & 31, wv = tid >> 5;

    // stage V (zero-padded past 49 keys)
    {
        int l = tid >> 1;
        int c0 = (tid & 1) * 16;
        v8h a = {}, bb = {};
        if (l < 49) {
            const _Float16* vp = kvbuf + (size_t)(b * 49 + l) * 768 + 384 + head * 32 + c0;
            a = *(const v8h*)vp; bb = *(const v8h*)(vp + 8);
        }
#pragma unroll
        for (int j = 0; j < 8; ++j) {
            Vt[(c0 + j) * 64 + l]     = a[j];
            Vt[(c0 + 8 + j) * 64 + l] = bb[j];
        }
    }
    __syncthreads();

    int m = lane & 15;
    int koA = (lane < 16) ? 0 : 8;
    int koB = (lane < 16) ? 0 : 8;

    const _Float16* qp = qbuf + (size_t)(qbase + wv * 16 + m) * DD + head * 32 + koA;
    v16h aQ = mk16(*(const v8h*)qp, *(const v8h*)(qp + 16));

    v8f s[4];
#pragma unroll
    for (int t = 0; t < 4; ++t)
#pragma unroll
        for (int i = 0; i < 8; ++i) s[t][i] = 0.f;

#pragma unroll
    for (int t = 0; t < 4; ++t) {
        int key = t * 16 + (lane & 15);
        v8h lo = {}, hi = {};
        if (key < 49) {
            const _Float16* kp = kvbuf + (size_t)(b * 49 + key) * 768 + head * 32 + koB;
            lo = *(const v8h*)kp; hi = *(const v8h*)(kp + 16);
        }
        s[t] = wmma16(aQ, mk16(lo, hi), s[t]);
    }

#pragma unroll
    for (int r = 0; r < 8; ++r) {
        float mx = -1e30f;
#pragma unroll
        for (int t = 0; t < 4; ++t) {
            int key = t * 16 + (lane & 15);
            float v = (key < 49) ? s[t][r] * SCALE_ : -1e30f;
            s[t][r] = v; mx = fmaxf(mx, v);
        }
#pragma unroll
        for (int off = 1; off < 16; off <<= 1) mx = fmaxf(mx, __shfl_xor(mx, off, 32));
        float sum = 0.f;
#pragma unroll
        for (int t = 0; t < 4; ++t) { float e = __expf(s[t][r] - mx); s[t][r] = e; sum += e; }
#pragma unroll
        for (int off = 1; off < 16; off <<= 1) sum += __shfl_xor(sum, off, 32);
        float inv = 1.f / sum;
        int rowl = r + ((lane < 16) ? 0 : 8);
#pragma unroll
        for (int t = 0; t < 4; ++t)
            Plds[wv * 1024 + rowl * 64 + t * 16 + (lane & 15)] = (_Float16)(s[t][r] * inv);
    }

    v8f o[2];
#pragma unroll
    for (int j = 0; j < 2; ++j)
#pragma unroll
        for (int i = 0; i < 8; ++i) o[j][i] = 0.f;
#pragma unroll
    for (int kc = 0; kc < 2; ++kc) {
        const _Float16* pp = Plds + wv * 1024 + m * 64 + kc * 32 + koA;
        v16h aP = mk16(*(const v8h*)pp, *(const v8h*)(pp + 16));
#pragma unroll
        for (int j = 0; j < 2; ++j) {
            int n = j * 16 + (lane & 15);
            const _Float16* vp = Vt + n * 64 + kc * 32 + koB;
            v16h bV = mk16(*(const v8h*)vp, *(const v8h*)(vp + 16));
            o[j] = wmma16(aP, bV, o[j]);
        }
    }
#pragma unroll
    for (int j = 0; j < 2; ++j)
#pragma unroll
        for (int r = 0; r < 8; ++r) {
            int q = qbase + wv * 16 + r + ((lane < 16) ? 0 : 8);
            out[(size_t)q * DD + head * 32 + j * 16 + (lane & 15)] = (_Float16)(o[j][r]);
        }
}

// ---------------------------------------------------------------------------
// im2col for the 8x8 stride-8 VALID conv (non-overlapping patches):
// dst[m, (ky*8+kx)*384 + c] = src[token(b,oy*8+ky,ox*8+kx), c]
// ---------------------------------------------------------------------------
__global__ __launch_bounds__(256) void im2col_kernel(const _Float16* __restrict__ src,
                                                     _Float16* __restrict__ dst) {
    int gid = blockIdx.x * 256 + threadIdx.x;      // 784 * 3072 chunks of 8
    if (gid >= 784 * 3072) return;
    int m  = gid / 3072;
    int c8 = gid % 3072;
    int k0 = c8 * 8;
    int p  = k0 / 384, cc = k0 % 384;
    int ky = p >> 3, kx = p & 7;
    int b = m / 49, rem = m % 49, oy = rem / 7, ox = rem % 7;
    int token = (b * HW_ + oy * 8 + ky) * HW_ + ox * 8 + kx;
    *(v8h*)&dst[(size_t)m * 24576 + k0] = *(const v8h*)&src[(size_t)token * DD + cc];
}

// ---------------------------------------------------------------------------
// Launch
// ---------------------------------------------------------------------------
extern "C" void kernel_launch(void* const* d_in, const int* in_sizes, int n_in,
                              void* d_out, int out_size, void* d_ws, size_t ws_size,
                              hipStream_t stream) {
    (void)in_sizes; (void)n_in; (void)out_size; (void)ws_size;

    const float* x     = (const float*)d_in[0];
    const float* g1    = (const float*)d_in[1];
    const float* be1   = (const float*)d_in[2];
    const float* w_qkv = (const float*)d_in[3];
    const float* b_qkv = (const float*)d_in[4];
    const float* w_lo  = (const float*)d_in[5];
    const float* b_lo  = (const float*)d_in[6];
    const float* g2    = (const float*)d_in[7];
    const float* be2   = (const float*)d_in[8];
    const float* w_f1l = (const float*)d_in[9];
    const float* b_f1l = (const float*)d_in[10];
    const float* w_f2l = (const float*)d_in[11];
    const float* b_f2l = (const float*)d_in[12];
    const float* g3    = (const float*)d_in[13];
    const float* be3   = (const float*)d_in[14];
    const float* w_q   = (const float*)d_in[15];
    const float* w_kv  = (const float*)d_in[16];
    const float* convw = (const float*)d_in[17];
    const float* convb = (const float*)d_in[18];
    const float* w_go  = (const float*)d_in[19];
    const float* b_go  = (const float*)d_in[20];
    const float* g4    = (const float*)d_in[21];
    const float* be4   = (const float*)d_in[22];
    const float* w_f1g = (const float*)d_in[23];
    const float* b_f1g = (const float*)d_in[24];
    const float* w_f2g = (const float*)d_in[25];
    const float* b_f2g = (const float*)d_in[26];
    float* out = (float*)d_out;

    // ---- workspace carve (all f16, 256B aligned) ----
    char* ws = (char*)d_ws;
    size_t off = 0;
    auto take = [&](size_t halves) -> _Float16* {
        _Float16* p = (_Float16*)(ws + off);
        off += (halves * sizeof(_Float16) + 255) & ~(size_t)255;
        return p;
    };
    _Float16* w_qkv16 = take(384 * 1152);
    _Float16* w_lo16  = take(384 * 384);
    _Float16* w_f1l16 = take(384 * 1536);
    _Float16* w_f2l16 = take(1536 * 384);
    _Float16* w_q16   = take(384 * 384);
    _Float16* w_kv16  = take(384 * 768);
    _Float16* conv16  = take(8 * 8 * 384 * 384);
    _Float16* w_go16  = take(384 * 384);
    _Float16* w_f1g16 = take(384 * 1536);
    _Float16* w_f2g16 = take(1536 * 384);
    _Float16* hA      = take((size_t)TOK * DD);      // LN output / global-attn output
    _Float16* hB      = take((size_t)TOK * FF_);     // qkv / q / ffn hidden
    _Float16* hC      = take((size_t)TOK * DD);      // local-attn out / im2col (784*24576)
    _Float16* rbuf    = take(784 * 384);
    _Float16* kvbuf   = take(784 * 768);

    auto cv = [&](const float* s, _Float16* d, int n) {
        cvt_kernel<<<(n + 255) / 256, 256, 0, stream>>>(s, d, n);
    };
    cv(w_qkv, w_qkv16, 384 * 1152);
    cv(w_lo,  w_lo16,  384 * 384);
    cv(w_f1l, w_f1l16, 384 * 1536);
    cv(w_f2l, w_f2l16, 1536 * 384);
    cv(w_q,   w_q16,   384 * 384);
    cv(w_kv,  w_kv16,  384 * 768);
    cv(convw, conv16,  8 * 8 * 384 * 384);
    cv(w_go,  w_go16,  384 * 384);
    cv(w_f1g, w_f1g16, 384 * 1536);
    cv(w_f2g, w_f2g16, 1536 * 384);

    const dim3 blk256(256), blk128(128);
    const int MT = (TOK + 127) / 128;   // 392
    const int MT2 = (784 + 127) / 128;  // 7

    // ---- block 1: local attention ----
    ln_kernel<<<TOK / 8, blk256, 0, stream>>>(x, g1, be1, hA, TOK);
    gemm_kernel<false><<<dim3(1152 / 128, MT), blk256, 0, stream>>>(
        hA, w_qkv16, b_qkv, nullptr, nullptr, hB, TOK, 1152, 384);
    local_attn_kernel<<<NB * 49 * NH_, blk128, 0, stream>>>(hB, hC);
    gemm_kernel<false><<<dim3(384 / 128, MT), blk256, 0, stream>>>(
        hC, w_lo16, b_lo, x, out, nullptr, TOK, 384, 384);

    // ---- block 2: FFN local ----
    ln_kernel<<<TOK / 8, blk256, 0, stream>>>(out, g2, be2, hA, TOK);
    gemm_kernel<true><<<dim3(1536 / 128, MT), blk256, 0, stream>>>(
        hA, w_f1l16, b_f1l, nullptr, nullptr, hB, TOK, 1536, 384);
    gemm_kernel<false><<<dim3(384 / 128, MT), blk256, 0, stream>>>(
        hB, w_f2l16, b_f2l, out, out, nullptr, TOK, 384, 1536);

    // ---- block 3: global attention ----
    ln_kernel<<<TOK / 8, blk256, 0, stream>>>(out, g3, be3, hA, TOK);
    gemm_kernel<false><<<dim3(384 / 128, MT), blk256, 0, stream>>>(
        hA, w_q16, nullptr, nullptr, nullptr, hB, TOK, 384, 384);            // q
    im2col_kernel<<<(784 * 3072 + 255) / 256, blk256, 0, stream>>>(hA, hC);
    gemm_kernel<false><<<dim3(384 / 128, MT2), blk256, 0, stream>>>(
        hC, conv16, convb, nullptr, nullptr, rbuf, 784, 384, 24576);         // conv as GEMM
    gemm_kernel<false><<<dim3(768 / 128, MT2), blk256, 0, stream>>>(
        rbuf, w_kv16, nullptr, nullptr, nullptr, kvbuf, 784, 768, 384);      // k|v
    global_attn_kernel<<<NB * NH_ * 49, blk128, 0, stream>>>(hB, kvbuf, hA);
    gemm_kernel<false><<<dim3(384 / 128, MT), blk256, 0, stream>>>(
        hA, w_go16, b_go, out, out, nullptr, TOK, 384, 384);

    // ---- block 4: FFN global ----
    ln_kernel<<<TOK / 8, blk256, 0, stream>>>(out, g4, be4, hA, TOK);
    gemm_kernel<true><<<dim3(1536 / 128, MT), blk256, 0, stream>>>(
        hA, w_f1g16, b_f1g, nullptr, nullptr, hB, TOK, 1536, 384);
    gemm_kernel<false><<<dim3(384 / 128, MT), blk256, 0, stream>>>(
        hB, w_f2g16, b_f2g, out, out, nullptr, TOK, 384, 1536);
}
